// TransitionUp_90185723281822
// MI455X (gfx1250) — matrix operand order repaired and checked by reading, not compiled, not run on hardware
//
#include <hip/hip_runtime.h>

typedef unsigned short u16;
typedef __bf16 bf16_t;
typedef bf16_t v16bf __attribute__((ext_vector_type(16)));
typedef bf16_t v8bf  __attribute__((ext_vector_type(8)));
typedef float  v8f   __attribute__((ext_vector_type(8)));
typedef int    v4i   __attribute__((vector_size(16)));

#define BB 8
#define N1P 1024
#define N2P 4096
#define C1F 512
#define C2F 256
#define FEA_IN 768
#define FEA_OUT 512
#define MROWS (BB * N2P)   // 32768

#if __has_builtin(__builtin_amdgcn_global_load_async_to_lds_b128) && \
    __has_builtin(__builtin_amdgcn_s_wait_asynccnt)
#define USE_ASYNC_LDS 1
#else
#define USE_ASYNC_LDS 0
#endif

// async builtin takes typed 16B-vector pointers: (global int4*, local int4*, imm off, imm cpol)
#define G1V4(p) ((__attribute__((address_space(1))) v4i*)(p))
#define L3V4(p) ((__attribute__((address_space(3))) v4i*)(p))

__device__ __forceinline__ u16 f2bf(float f) {
  union { float f; unsigned u; } x; x.f = f;
  unsigned r = (x.u + 0x7FFFu + ((x.u >> 16) & 1u)) >> 16;
  return (u16)r;
}

// ---------------- weight conversion f32 -> bf16 (layout preserved) ----------
__global__ __launch_bounds__(256) void k_f32_to_bf16(const float* __restrict__ src,
                                                     u16* __restrict__ dst, int n) {
  int i = blockIdx.x * 256 + threadIdx.x;
  if (i < n) dst[i] = f2bf(src[i]);
}

// ---------------- 3-NN search (p1 tile in LDS) + p2 passthrough -------------
__global__ __launch_bounds__(256) void k_knn(const float* __restrict__ p1,
                                             const float* __restrict__ p2,
                                             int* __restrict__ idx_ws,
                                             float* __restrict__ w_ws,
                                             float* __restrict__ p2_out) {
  __shared__ float sp1[N1P * 3];
  const int tid = threadIdx.x;
  const int gid = blockIdx.x * 256 + tid;        // point id 0..32767
  const int b = gid >> 12;                       // batch (4096 pts/batch)
  for (int i = tid; i < N1P * 3; i += 256) sp1[i] = p1[b * N1P * 3 + i];
  __syncthreads();
  const float px = p2[gid * 3 + 0], py = p2[gid * 3 + 1], pz = p2[gid * 3 + 2];
  float d0 = 3.4e38f, d1 = 3.4e38f, d2 = 3.4e38f;
  int i0 = 0, i1 = 0, i2 = 0;
  for (int j = 0; j < N1P; ++j) {
    float dx = px - sp1[j * 3 + 0];
    float dy = py - sp1[j * 3 + 1];
    float dz = pz - sp1[j * 3 + 2];
    float d = dx * dx + dy * dy + dz * dz;
    if (d < d0)      { d2 = d1; i2 = i1; d1 = d0; i1 = i0; d0 = d; i0 = j; }
    else if (d < d1) { d2 = d1; i2 = i1; d1 = d;  i1 = j; }
    else if (d < d2) { d2 = d;  i2 = j; }
  }
  float r0 = 1.f / (sqrtf(d0) + 1e-8f);
  float r1 = 1.f / (sqrtf(d1) + 1e-8f);
  float r2 = 1.f / (sqrtf(d2) + 1e-8f);
  float rs = 1.f / (r0 + r1 + r2);
  idx_ws[gid * 3 + 0] = i0; idx_ws[gid * 3 + 1] = i1; idx_ws[gid * 3 + 2] = i2;
  w_ws[gid * 3 + 0] = r0 * rs; w_ws[gid * 3 + 1] = r1 * rs; w_ws[gid * 3 + 2] = r2 * rs;
  p2_out[gid * 3 + 0] = px; p2_out[gid * 3 + 1] = py; p2_out[gid * 3 + 2] = pz;
}

// ---------------- interpolate + concat -> y bf16 [32768 x 768] --------------
__global__ __launch_bounds__(256) void k_interp_concat(const float* __restrict__ x1,
                                                       const float* __restrict__ x2,
                                                       const int* __restrict__ idx_ws,
                                                       const float* __restrict__ w_ws,
                                                       u16* __restrict__ y) {
  const int gid = blockIdx.x;          // point id
  const int tid = threadIdx.x;
  const int b = gid >> 12;
  const int i0 = idx_ws[gid * 3 + 0], i1 = idx_ws[gid * 3 + 1], i2 = idx_ws[gid * 3 + 2];
  const float w0 = w_ws[gid * 3 + 0], w1 = w_ws[gid * 3 + 1], w2 = w_ws[gid * 3 + 2];
  const float* r0 = x1 + (size_t)(b * N1P + i0) * C1F;
  const float* r1 = x1 + (size_t)(b * N1P + i1) * C1F;
  const float* r2 = x1 + (size_t)(b * N1P + i2) * C1F;
  u16* yr = y + (size_t)gid * FEA_IN;
#pragma unroll
  for (int s = 0; s < 2; ++s) {
    int c = tid + s * 256;
    yr[c] = f2bf(w0 * r0[c] + w1 * r1[c] + w2 * r2[c]);
  }
  yr[C1F + tid] = f2bf(x2[(size_t)gid * C2F + tid]);
}

// ---------------- bf16 WMMA GEMM: C[M,N] = A[M,K] * Bw[N,K]^T + bias --------
// block: 256 thr (8 waves); tile 128(M) x 64(N), BK=32; wave -> 32x32 subtile.
// Async path: double-buffered LDS fed by GLOBAL_LOAD_ASYNC_TO_LDS_B128,
// one barrier per K-step, next tile in flight during current tile's WMMAs.
__global__ __launch_bounds__(256) void k_gemm_bf16_wmma(const u16* __restrict__ A,
                                                        const u16* __restrict__ Bw,
                                                        const float* __restrict__ bias,
                                                        float* __restrict__ C,
                                                        int K, int N) {
  __shared__ u16 As[2][128 * 32] __attribute__((aligned(64)));
  __shared__ u16 Bs[2][64 * 32]  __attribute__((aligned(64)));
  const int tid = threadIdx.x;
  const int m0 = blockIdx.x * 128;
  const int n0 = blockIdx.y * 64;
  const int wave = tid >> 5, lane = tid & 31;
  const int wm = wave & 3, wn = wave >> 2;
  const int half = lane >> 4, l16 = lane & 15;

  // global staging: A 16 bf16/thr (2 x 16B), B 8 bf16/thr (1 x 16B)
  const int ar = tid >> 1, ac = (tid & 1) * 16;
  const int br = tid >> 2, bc = (tid & 3) * 8;
  const u16* aG = A + (size_t)(m0 + ar) * K + ac;
  const u16* bG = Bw + (size_t)(n0 + br) * K + bc;
  const int aSo = ar * 32 + ac;
  const int bSo = br * 32 + bc;

  // LDS fragment offsets: contiguous k order (k = 16*half + s) for both A and B
  const int aoff = (wm * 32 + l16) * 32 + half * 16;    // row l16, K half per lane-half
  const int boff = (wn * 32 + l16) * 32 + half * 16;    // col l16, K half per lane-half

  v8f acc00 = {}, acc01 = {}, acc10 = {}, acc11 = {};

#if USE_ASYNC_LDS
  // prologue: async-stage first K tile into buffer 0
  __builtin_amdgcn_global_load_async_to_lds_b128(G1V4(aG), L3V4(&As[0][aSo]), 0, 0);
  __builtin_amdgcn_global_load_async_to_lds_b128(G1V4(aG + 8), L3V4(&As[0][aSo + 8]), 0, 0);
  __builtin_amdgcn_global_load_async_to_lds_b128(G1V4(bG), L3V4(&Bs[0][bSo]), 0, 0);
  int buf = 0;
#pragma unroll 2
  for (int k0 = 0; k0 < K; k0 += 32) {
    __builtin_amdgcn_s_wait_asynccnt(0);   // my tile for `buf` has landed in LDS
    __syncthreads();                       // everyone's tile landed; other buf free
    if (k0 + 32 < K) {
      const u16* aN = aG + k0 + 32;
      const u16* bN = bG + k0 + 32;
      __builtin_amdgcn_global_load_async_to_lds_b128(G1V4(aN), L3V4(&As[buf ^ 1][aSo]), 0, 0);
      __builtin_amdgcn_global_load_async_to_lds_b128(G1V4(aN + 8), L3V4(&As[buf ^ 1][aSo + 8]), 0, 0);
      __builtin_amdgcn_global_load_async_to_lds_b128(G1V4(bN), L3V4(&Bs[buf ^ 1][bSo]), 0, 0);
    }
    v8bf a0lo = *reinterpret_cast<const v8bf*>(&As[buf][aoff]);
    v8bf a0hi = *reinterpret_cast<const v8bf*>(&As[buf][aoff + 8]);
    v8bf a1lo = *reinterpret_cast<const v8bf*>(&As[buf][aoff + 16 * 32]);
    v8bf a1hi = *reinterpret_cast<const v8bf*>(&As[buf][aoff + 16 * 32 + 8]);
    v16bf a0 = __builtin_shufflevector(a0lo, a0hi, 0,1,2,3,4,5,6,7,8,9,10,11,12,13,14,15);
    v16bf a1 = __builtin_shufflevector(a1lo, a1hi, 0,1,2,3,4,5,6,7,8,9,10,11,12,13,14,15);
    v16bf b0 = *reinterpret_cast<const v16bf*>(&Bs[buf][boff]);
    v16bf b1 = *reinterpret_cast<const v16bf*>(&Bs[buf][boff + 16 * 32]);

    acc00 = __builtin_amdgcn_wmma_f32_16x16x32_bf16(false, a0, false, b0, (short)0, acc00, false, false);
    acc01 = __builtin_amdgcn_wmma_f32_16x16x32_bf16(false, a0, false, b1, (short)0, acc01, false, false);
    acc10 = __builtin_amdgcn_wmma_f32_16x16x32_bf16(false, a1, false, b0, (short)0, acc10, false, false);
    acc11 = __builtin_amdgcn_wmma_f32_16x16x32_bf16(false, a1, false, b1, (short)0, acc11, false, false);
    buf ^= 1;
  }
#else
  // fallback: VGPR-staged LDS, two barriers per K-step
  for (int k0 = 0; k0 < K; k0 += 32) {
    const uint4* ag = reinterpret_cast<const uint4*>(aG + k0);
    uint4 av0 = ag[0];
    uint4 av1 = ag[1];
    uint4 bv = *reinterpret_cast<const uint4*>(bG + k0);
    if (k0 + 32 < K) {
      __builtin_prefetch(aG + k0 + 32, 0, 1);
      __builtin_prefetch(bG + k0 + 32, 0, 1);
    }
    __syncthreads();
    reinterpret_cast<uint4*>(&As[0][aSo])[0] = av0;
    reinterpret_cast<uint4*>(&As[0][aSo])[1] = av1;
    *reinterpret_cast<uint4*>(&Bs[0][bSo]) = bv;
    __syncthreads();

    v8bf a0lo = *reinterpret_cast<const v8bf*>(&As[0][aoff]);
    v8bf a0hi = *reinterpret_cast<const v8bf*>(&As[0][aoff + 8]);
    v8bf a1lo = *reinterpret_cast<const v8bf*>(&As[0][aoff + 16 * 32]);
    v8bf a1hi = *reinterpret_cast<const v8bf*>(&As[0][aoff + 16 * 32 + 8]);
    v16bf a0 = __builtin_shufflevector(a0lo, a0hi, 0,1,2,3,4,5,6,7,8,9,10,11,12,13,14,15);
    v16bf a1 = __builtin_shufflevector(a1lo, a1hi, 0,1,2,3,4,5,6,7,8,9,10,11,12,13,14,15);
    v16bf b0 = *reinterpret_cast<const v16bf*>(&Bs[0][boff]);
    v16bf b1 = *reinterpret_cast<const v16bf*>(&Bs[0][boff + 16 * 32]);

    acc00 = __builtin_amdgcn_wmma_f32_16x16x32_bf16(false, a0, false, b0, (short)0, acc00, false, false);
    acc01 = __builtin_amdgcn_wmma_f32_16x16x32_bf16(false, a0, false, b1, (short)0, acc01, false, false);
    acc10 = __builtin_amdgcn_wmma_f32_16x16x32_bf16(false, a1, false, b0, (short)0, acc10, false, false);
    acc11 = __builtin_amdgcn_wmma_f32_16x16x32_bf16(false, a1, false, b1, (short)0, acc11, false, false);
  }
#endif

  // epilogue: 16x16 f32 C/D layout (lane col = l16, VGPR v -> row v + half*8)
  const int col0 = n0 + wn * 32 + l16;
  const int col1 = col0 + 16;
  const float bia0 = bias[col0], bia1 = bias[col1];
  const int r0b = m0 + wm * 32 + half * 8;
#pragma unroll
  for (int v = 0; v < 8; ++v) {
    const size_t r = (size_t)(r0b + v);
    C[r * N + col0]        = acc00[v] + bia0;
    C[r * N + col1]        = acc01[v] + bia1;
    C[(r + 16) * N + col0] = acc10[v] + bia0;
    C[(r + 16) * N + col1] = acc11[v] + bia1;
  }
}

// ---------------- BN: deterministic two-stage column reduction --------------
__global__ __launch_bounds__(256) void k_bn_partial(const float* __restrict__ z,
                                                    float* __restrict__ partS,
                                                    float* __restrict__ partQ) {
  const int tid = threadIdx.x;
  const int bk = blockIdx.x;              // 512 blocks x 64 rows
  const int ch0 = tid, ch1 = tid + 256;
  float s0 = 0, s1 = 0, q0 = 0, q1 = 0;
  const float* base = z + (size_t)bk * 64 * FEA_OUT;
  for (int r = 0; r < 64; ++r) {
    float v0 = base[r * FEA_OUT + ch0];
    float v1 = base[r * FEA_OUT + ch1];
    s0 += v0; q0 += v0 * v0;
    s1 += v1; q1 += v1 * v1;
  }
  partS[bk * FEA_OUT + ch0] = s0; partS[bk * FEA_OUT + ch1] = s1;
  partQ[bk * FEA_OUT + ch0] = q0; partQ[bk * FEA_OUT + ch1] = q1;
}

__global__ __launch_bounds__(256) void k_bn_finalize(const float* __restrict__ partS,
                                                     const float* __restrict__ partQ,
                                                     const float* __restrict__ gamma,
                                                     const float* __restrict__ beta,
                                                     float* __restrict__ ca,
                                                     float* __restrict__ cb) {
  const int ch = blockIdx.x * 256 + threadIdx.x;   // 0..511
  float s = 0, q = 0;
  for (int p = 0; p < 512; ++p) {
    s += partS[p * FEA_OUT + ch];
    q += partQ[p * FEA_OUT + ch];
  }
  const float invN = 1.0f / (float)MROWS;
  float mean = s * invN;
  float var = q * invN - mean * mean;
  float inv = rsqrtf(var + 1e-5f);
  float a = inv * gamma[ch];
  ca[ch] = a;
  cb[ch] = beta[ch] - mean * a;
}

__global__ __launch_bounds__(256) void k_bn_relu_bf16(const float* __restrict__ z,
                                                      const float* __restrict__ ca,
                                                      const float* __restrict__ cb,
                                                      u16* __restrict__ out) {
  const size_t i0 = ((size_t)blockIdx.x * 256 + threadIdx.x) * 4;
  const int ch = (int)(i0 & (FEA_OUT - 1));
  float4 v = *reinterpret_cast<const float4*>(z + i0);
  union { u16 h[4]; uint2 u; } pk;
  pk.h[0] = f2bf(fmaxf(v.x * ca[ch + 0] + cb[ch + 0], 0.f));
  pk.h[1] = f2bf(fmaxf(v.y * ca[ch + 1] + cb[ch + 1], 0.f));
  pk.h[2] = f2bf(fmaxf(v.z * ca[ch + 2] + cb[ch + 2], 0.f));
  pk.h[3] = f2bf(fmaxf(v.w * ca[ch + 3] + cb[ch + 3], 0.f));
  *reinterpret_cast<uint2*>(out + i0) = pk.u;
}

__global__ __launch_bounds__(256) void k_bn_f32_inplace(float* __restrict__ z,
                                                        const float* __restrict__ ca,
                                                        const float* __restrict__ cb) {
  const size_t i0 = ((size_t)blockIdx.x * 256 + threadIdx.x) * 4;
  const int ch = (int)(i0 & (FEA_OUT - 1));
  float4 v = *reinterpret_cast<float4*>(z + i0);
  v.x = v.x * ca[ch + 0] + cb[ch + 0];
  v.y = v.y * ca[ch + 1] + cb[ch + 1];
  v.z = v.z * ca[ch + 2] + cb[ch + 2];
  v.w = v.w * ca[ch + 3] + cb[ch + 3];
  *reinterpret_cast<float4*>(z + i0) = v;
}

// ---------------------------------------------------------------------------
extern "C" void kernel_launch(void* const* d_in, const int* in_sizes, int n_in,
                              void* d_out, int out_size, void* d_ws, size_t ws_size,
                              hipStream_t stream) {
  const float* x1  = (const float*)d_in[0];
  const float* p1  = (const float*)d_in[1];
  const float* x2  = (const float*)d_in[2];
  const float* p2  = (const float*)d_in[3];
  const float* W1  = (const float*)d_in[4];
  const float* b1  = (const float*)d_in[5];
  const float* g1  = (const float*)d_in[6];
  const float* be1 = (const float*)d_in[7];
  const float* W2  = (const float*)d_in[8];
  const float* b2  = (const float*)d_in[9];
  const float* g2  = (const float*)d_in[10];
  const float* be2 = (const float*)d_in[11];
  (void)in_sizes; (void)n_in; (void)out_size; (void)ws_size;

  char* ws = (char*)d_ws;
  int*   idx_ws = (int*)(ws + 0);                    //  384 KB
  float* w_ws   = (float*)(ws + 393216);             //  384 KB
  u16*   Wb1    = (u16*)(ws + 786432);               //  768 KB  [512,768] bf16
  u16*   Wb2    = (u16*)(ws + 1572864);              //  512 KB  [512,512] bf16
  float* partS  = (float*)(ws + 2097152);            //    1 MB
  float* partQ  = (float*)(ws + 3145728);            //    1 MB
  float* ca     = (float*)(ws + 4194304);            //    2 KB
  float* cb     = (float*)(ws + 4196352);            //    2 KB
  u16*   ybuf   = (u16*)(ws + 4202496);              //   48 MB  [32768,768] bf16
  float* z1     = (float*)(ws + 54534144);           //   64 MB  [32768,512] f32
  u16*   z1n    = ybuf;                              // alias (y dead after GEMM1)

  float* zout  = (float*)d_out;                      // [32768,512] f32
  float* p2out = zout + (size_t)MROWS * FEA_OUT;     // [32768,3]  f32

  k_f32_to_bf16<<<dim3((FEA_OUT * FEA_IN + 255) / 256), dim3(256), 0, stream>>>(
      W1, Wb1, FEA_OUT * FEA_IN);
  k_f32_to_bf16<<<dim3((FEA_OUT * FEA_OUT + 255) / 256), dim3(256), 0, stream>>>(
      W2, Wb2, FEA_OUT * FEA_OUT);

  k_knn<<<dim3(MROWS / 256), dim3(256), 0, stream>>>(p1, p2, idx_ws, w_ws, p2out);
  k_interp_concat<<<dim3(MROWS), dim3(256), 0, stream>>>(x1, x2, idx_ws, w_ws, ybuf);

  // layer 1: y[32768,768] @ W1^T -> z1 (pre-BN, f32)
  k_gemm_bf16_wmma<<<dim3(MROWS / 128, FEA_OUT / 64), dim3(256), 0, stream>>>(
      ybuf, Wb1, b1, z1, FEA_IN, FEA_OUT);
  k_bn_partial<<<dim3(MROWS / 64), dim3(256), 0, stream>>>(z1, partS, partQ);
  k_bn_finalize<<<dim3(2), dim3(256), 0, stream>>>(partS, partQ, g1, be1, ca, cb);
  k_bn_relu_bf16<<<dim3((MROWS * FEA_OUT / 4) / 256), dim3(256), 0, stream>>>(
      z1, ca, cb, z1n);

  // layer 2: z1n[32768,512] @ W2^T -> d_out (pre-BN, f32), then BN in place
  k_gemm_bf16_wmma<<<dim3(MROWS / 128, FEA_OUT / 64), dim3(256), 0, stream>>>(
      z1n, Wb2, b2, zout, FEA_OUT, FEA_OUT);
  k_bn_partial<<<dim3(MROWS / 64), dim3(256), 0, stream>>>(zout, partS, partQ);
  k_bn_finalize<<<dim3(2), dim3(256), 0, stream>>>(partS, partQ, g2, be2, ca, cb);
  k_bn_f32_inplace<<<dim3((MROWS * FEA_OUT / 4) / 256), dim3(256), 0, stream>>>(
      zout, ca, cb);
}
